// MultiHeadDotProductAttention_43001212568068
// MI455X (gfx1250) — matrix operand order repaired and checked by reading, not compiled, not run on hardware
//
#include <hip/hip_runtime.h>
#include <hip/hip_bf16.h>

#define B_ 2
#define L_ 2048
#define E_ 2048
#define H_ 16
#define D_ 128

typedef __attribute__((ext_vector_type(16))) __bf16 v16bf;
typedef __attribute__((ext_vector_type(8)))  __bf16 v8bf;
typedef __attribute__((ext_vector_type(8)))  float  v8f;

__device__ __forceinline__ v8f wmma_bf16(v16bf a, v16bf b, v8f c) {
  // 8 args: (neg_a, A, neg_b, B, c_mod, C, reuse_a, reuse_b)
  return __builtin_amdgcn_wmma_f32_16x16x32_bf16(false, a, false, b, (short)0, c,
                                                 false, false);
}

// Issue two async global->LDS 16B transfers (per lane). Tracked by ASYNCcnt.
__device__ __forceinline__ void async_g2l_32B(uint32_t lds, uint64_t gaddr) {
  asm volatile("global_load_async_to_lds_b128 %0, %1, off\n\t"
               "global_load_async_to_lds_b128 %2, %3, off"
               :: "v"(lds), "v"(gaddr), "v"(lds + 16), "v"(gaddr + 16)
               : "memory");
}

// ---------------- fp32 -> bf16 conversion ----------------
__global__ void k_f32_to_bf16(const float* __restrict__ src,
                              __bf16* __restrict__ dst, int n) {
  int i = (blockIdx.x * blockDim.x + threadIdx.x) * 4;
  if (i >= n) return;
  float4 f = *(const float4*)(src + i);
  dst[i + 0] = (__bf16)f.x;
  dst[i + 1] = (__bf16)f.y;
  dst[i + 2] = (__bf16)f.z;
  dst[i + 3] = (__bf16)f.w;
}

// ---------------- GEMM: C[M,N] = A[M,K] * W[N,K]^T ----------------
// Block = 4 waves = 64x64 C tile. Weight panel (64 cols x 32 k, bf16) is
// async-DMA'd to LDS (double buffered) and shared by all 4 waves; each wave
// owns 16 rows and keeps its A fragments in registers. Requires M%64==N%64==0,
// K%32==0; grid covers tiles exactly (no early exit -> barriers are uniform).
template <bool OUTF32>
__global__ void __launch_bounds__(128, 1)
k_gemm(const __bf16* __restrict__ A, const __bf16* __restrict__ W,
       float* __restrict__ Cf, __bf16* __restrict__ Cb, int M, int N, int K) {
  __shared__ __bf16 Wp[2][64 * 32];      // [buf][n][k] 2 x 4KB

  const int lane = threadIdx.x & 31;
  const int wave = threadIdx.x >> 5;
  const int half = lane >> 4;
  const int l16  = lane & 15;
  const int tilesN = N >> 6;
  const int tm = blockIdx.x / tilesN;
  const int tn = blockIdx.x % tilesN;

  // This thread's slice of the panel loader: row pn, 32B half `pof`.
  const int pn  = threadIdx.x >> 1;            // 0..63  (C column)
  const int pof = (threadIdx.x & 1) * 32;      // byte offset in 64B row
  const uint64_t wsrc = (uint64_t)(uintptr_t)(W + (size_t)(tn * 64 + pn) * K) + pof;
  const uint32_t wlds[2] = {
      (uint32_t)(uintptr_t)(&Wp[0][pn * 32]) + (uint32_t)pof,
      (uint32_t)(uintptr_t)(&Wp[1][pn * 32]) + (uint32_t)pof };

  const __bf16* Arow = A + (size_t)(tm * 64 + wave * 16 + l16) * K;

  // Prime buffer 0 with the k=0 panel.
  async_g2l_32B(wlds[0], wsrc);

  v8f acc[4] = {};
  for (int k0 = 0; k0 < K; k0 += 32) {
    const int cur = (k0 >> 5) & 1;
    const int k0n = (k0 + 32 < K) ? (k0 + 32) : 0;   // wrap keeps wait-count uniform
    __syncthreads();                                  // buf cur^1 fully consumed
    async_g2l_32B(wlds[cur ^ 1], wsrc + (size_t)k0n * 2);
    // In-order async completion: <=2 outstanding means cur panel has landed.
    asm volatile("s_wait_asynccnt 0x2" ::: "memory");
    __syncthreads();                                  // whole cur panel visible

    // A 16x32 fragment: lanes 0-15 K 0..15, lanes 16-31 K 16..31
    const int ka = k0 + half * 16;
    v8bf alo = *(const v8bf*)(Arow + ka);
    v8bf ahi = *(const v8bf*)(Arow + ka + 8);
    v16bf a;
#pragma unroll
    for (int i = 0; i < 8; ++i) { a[i] = alo[i]; a[8 + i] = ahi[i]; }

    const __bf16* wp = Wp[cur];
#pragma unroll
    for (int j = 0; j < 4; ++j) {
      // B 32x16 fragment: lane = column, lanes 0-15 K 0..15, 16-31 K 16..31
      v16bf b = *(const v16bf*)(wp + (j * 16 + l16) * 32 + half * 16);
      acc[j] = wmma_bf16(a, b, acc[j]);
    }
  }

  // C/D layout: VGPR r -> row r (lanes 0-15) / r+8 (lanes 16-31), col = lane%16
#pragma unroll
  for (int j = 0; j < 4; ++j) {
#pragma unroll
    for (int r = 0; r < 8; ++r) {
      const size_t m = (size_t)(tm * 64 + wave * 16 + r + half * 8);
      const size_t n = (size_t)(tn * 64 + j * 16 + l16);
      if (OUTF32) Cf[m * N + n] = acc[j][r];
      else        Cb[m * N + n] = (__bf16)acc[j][r];
    }
  }
}

// ---------------- RMSNorm + rotary (+ optional sqrt(D) fold) ----------------
// One wave per (b,l,h) row of D=128; fp32 math; in-place on bf16 tensor.
__global__ void k_rms_rope(__bf16* __restrict__ x, const float* __restrict__ rope,
                           const float* __restrict__ scale, float mul) {
  const int lane = threadIdx.x & 31;
  const int wave = threadIdx.x >> 5;
  const int row = blockIdx.x * 4 + wave;          // flat (b,l,h)
  const int b = row / (L_ * H_);
  const int l = (row / H_) % L_;
  __bf16* xr = x + (size_t)row * D_;
  const float* rp = rope + ((size_t)b * L_ + l) * D_;   // [D/2][2] per (b,l)
  const int d0 = lane * 4;

  float xv[4];
  float ss = 0.f;
#pragma unroll
  for (int j = 0; j < 4; ++j) {
    xv[j] = (float)xr[d0 + j];
    ss += xv[j] * xv[j];
  }
#pragma unroll
  for (int off = 16; off; off >>= 1) ss += __shfl_xor(ss, off, 32);
  const float inv = rsqrtf(ss * (1.0f / D_) + 1e-6f);

  float xn[4];
#pragma unroll
  for (int j = 0; j < 4; ++j) xn[j] = scale[d0 + j] * xv[j] * inv;
#pragma unroll
  for (int p = 0; p < 2; ++p) {
    const float c = rp[(d0 / 2 + p) * 2 + 0];
    const float s = rp[(d0 / 2 + p) * 2 + 1];
    const float a0 = xn[2 * p], a1 = xn[2 * p + 1];
    xr[d0 + 2 * p + 0] = (__bf16)((a0 * c - a1 * s) * mul);
    xr[d0 + 2 * p + 1] = (__bf16)((a1 * c + a0 * s) * mul);
  }
}

// ---------------- Flash attention core ----------------
// Block = 4 waves, all same (b,h); shares 32-key K/V tiles through LDS.
// K tile streams in via async global->LDS DMA; V tile is transposed through
// registers (async cannot transpose). Online softmax; O in fp32 registers.
__global__ void __launch_bounds__(128, 1) k_flash(
    const __bf16* __restrict__ Q, const __bf16* __restrict__ Kg,
    const __bf16* __restrict__ Vg, const float* __restrict__ bias,
    __bf16* __restrict__ O) {
  __shared__ __bf16 Ks[32 * 128];        // [key][d]
  __shared__ __bf16 Vs[128 * 32];        // [d][key] (transposed for PV B-frags)
  __shared__ __bf16 Ps[4][16 * 32];      // per-wave P staging (C-layout -> A-layout)

  const int lane = threadIdx.x & 31;
  const int wave = threadIdx.x >> 5;
  const int half = lane >> 4;
  const int l16  = lane & 15;

  const int qblk = blockIdx.x % (L_ / 64);
  const int bh   = blockIdx.x / (L_ / 64);
  const int b = bh / H_;
  const int h = bh % H_;
  const int qt = qblk * 4 + wave;        // 16-row query tile

  // Preload Q as 4 A-fragments (16x32 chunks over D=128).
  const __bf16* Qrow = Q + ((size_t)(b * L_ + qt * 16 + l16) * H_ + h) * D_;
  v16bf aq[4];
#pragma unroll
  for (int c = 0; c < 4; ++c) {
    const int ka = c * 32 + half * 16;
    v8bf lo = *(const v8bf*)(Qrow + ka);
    v8bf hi = *(const v8bf*)(Qrow + ka + 8);
#pragma unroll
    for (int i = 0; i < 8; ++i) { aq[c][i] = lo[i]; aq[c][8 + i] = hi[i]; }
  }

  v8f o[8] = {};
  float mrow[8], lrow[8];
#pragma unroll
  for (int r = 0; r < 8; ++r) { mrow[r] = -3.0e38f; lrow[r] = 0.f; }

  const int crow = threadIdx.x >> 2;     // 0..31: key row of coop load
  const int cseg = threadIdx.x & 3;      // 0..3 : 32-element d segment
  const uint32_t klds = (uint32_t)(uintptr_t)(Ks + crow * 128 + cseg * 32);

  for (int kc = 0; kc < L_; kc += 32) {
    __syncthreads();                     // previous tile fully consumed
    {
      const size_t src = ((size_t)(b * L_ + kc + crow) * H_ + h) * D_ + cseg * 32;
      // K tile: async DMA straight into LDS (64B per thread).
      const uint64_t kg = (uint64_t)(uintptr_t)(Kg + src);
      async_g2l_32B(klds, kg);
      async_g2l_32B(klds + 32, kg + 32);
      // V tile: load to registers, scatter transposed into LDS.
      const uint4* vs4 = (const uint4*)(Vg + src);
      alignas(16) __bf16 tmp[32];
      uint4* t4 = (uint4*)tmp;
      t4[0] = vs4[0]; t4[1] = vs4[1]; t4[2] = vs4[2]; t4[3] = vs4[3];
#pragma unroll
      for (int j = 0; j < 32; ++j) Vs[(cseg * 32 + j) * 32 + crow] = tmp[j];
    }
    asm volatile("s_wait_asynccnt 0x0" ::: "memory");
    __syncthreads();

    // S = Q*K^T over 32 keys -> two 16x16 fp32 tiles
    v8f s0 = {}, s1 = {};
#pragma unroll
    for (int c = 0; c < 4; ++c) {
      const int kb = c * 32 + half * 16;
      v16bf b0 = *(const v16bf*)(Ks + l16 * 128 + kb);
      v16bf b1 = *(const v16bf*)(Ks + (16 + l16) * 128 + kb);
      s0 = wmma_bf16(aq[c], b0, s0);
      s1 = wmma_bf16(aq[c], b1, s1);
    }
    // + bias[b,0,m,k]
    const float* bp = bias + ((size_t)b * L_ + qt * 16 + half * 8) * L_ + kc;
#pragma unroll
    for (int r = 0; r < 8; ++r) {
      s0[r] += bp[(size_t)r * L_ + l16];
      s1[r] += bp[(size_t)r * L_ + 16 + l16];
    }

    // Online softmax (row stats replicated across each 16-lane group)
    float scl[8];
#pragma unroll
    for (int r = 0; r < 8; ++r) {
      float mx = fmaxf(s0[r], s1[r]);
#pragma unroll
      for (int off = 8; off; off >>= 1) mx = fmaxf(mx, __shfl_xor(mx, off, 32));
      const float mnew = fmaxf(mrow[r], mx);
      const float sc = __expf(mrow[r] - mnew);
      const float p0 = __expf(s0[r] - mnew);
      const float p1 = __expf(s1[r] - mnew);
      float rs = p0 + p1;
#pragma unroll
      for (int off = 8; off; off >>= 1) rs += __shfl_xor(rs, off, 32);
      lrow[r] = lrow[r] * sc + rs;
      mrow[r] = mnew;
      scl[r] = sc;
      Ps[wave][(r + half * 8) * 32 + l16]      = (__bf16)p0;
      Ps[wave][(r + half * 8) * 32 + 16 + l16] = (__bf16)p1;
    }
#pragma unroll
    for (int dt = 0; dt < 8; ++dt)
#pragma unroll
      for (int r = 0; r < 8; ++r) o[dt][r] *= scl[r];

    // LDS is in-order per wave; wait for the P stores before cross-lane readback
    asm volatile("s_wait_dscnt 0x0" ::: "memory");

    // Re-read P in A-matrix 16x32 layout
    v16bf pa;
    {
      const __bf16* pr = &Ps[wave][l16 * 32 + half * 16];
#pragma unroll
      for (int i = 0; i < 8; ++i) { pa[i] = pr[i]; pa[8 + i] = pr[8 + i]; }
    }
    // O += P*V  (V^T rows contiguous in Vs)
#pragma unroll
    for (int dt = 0; dt < 8; ++dt) {
      v16bf vbf = *(const v16bf*)(Vs + (dt * 16 + l16) * 32 + half * 16);
      o[dt] = wmma_bf16(pa, vbf, o[dt]);
    }
  }

  // O / l -> [B,L,H,D] bf16
#pragma unroll
  for (int dt = 0; dt < 8; ++dt) {
#pragma unroll
    for (int r = 0; r < 8; ++r) {
      const size_t m = (size_t)(qt * 16 + r + half * 8);
      O[((size_t)(b * L_ + m) * H_ + h) * D_ + dt * 16 + l16] =
          (__bf16)(o[dt][r] / lrow[r]);
    }
  }
}

extern "C" void kernel_launch(void* const* d_in, const int* in_sizes, int n_in,
                              void* d_out, int out_size, void* d_ws, size_t ws_size,
                              hipStream_t stream) {
  (void)in_sizes; (void)n_in; (void)out_size; (void)ws_size;
  const float* inputs_q  = (const float*)d_in[0];
  const float* inputs_kv = (const float*)d_in[1];
  const float* bias      = (const float*)d_in[2];
  const float* rope_q    = (const float*)d_in[3];
  const float* rope_k    = (const float*)d_in[4];
  const float* Wq        = (const float*)d_in[5];
  const float* Wk        = (const float*)d_in[6];
  const float* Wv        = (const float*)d_in[7];
  const float* Wo        = (const float*)d_in[8];
  const float* q_scale   = (const float*)d_in[9];
  const float* k_scale   = (const float*)d_in[10];
  float* out = (float*)d_out;

  const int BLE = B_ * L_ * E_;   // 8388608
  const int EE  = E_ * E_;        // 4194304
  __bf16* p  = (__bf16*)d_ws;
  __bf16* xq  = p; p += BLE;
  __bf16* xkv = p; p += BLE;
  __bf16* wq  = p; p += EE;
  __bf16* wk  = p; p += EE;
  __bf16* wv  = p; p += EE;
  __bf16* wo  = p; p += EE;
  __bf16* qb  = p; p += BLE;
  __bf16* kb  = p; p += BLE;
  __bf16* vb  = p; p += BLE;
  __bf16* ab  = p; p += BLE;

  // 1) fp32 -> bf16 staging
  k_f32_to_bf16<<<BLE / 1024, 256, 0, stream>>>(inputs_q,  xq,  BLE);
  k_f32_to_bf16<<<BLE / 1024, 256, 0, stream>>>(inputs_kv, xkv, BLE);
  k_f32_to_bf16<<<EE  / 1024, 256, 0, stream>>>(Wq, wq, EE);
  k_f32_to_bf16<<<EE  / 1024, 256, 0, stream>>>(Wk, wk, EE);
  k_f32_to_bf16<<<EE  / 1024, 256, 0, stream>>>(Wv, wv, EE);
  k_f32_to_bf16<<<EE  / 1024, 256, 0, stream>>>(Wo, wo, EE);

  // 2) Q/K/V projections (WMMA bf16, fp32 accum, async-LDS weight panels)
  const int M = B_ * L_;
  const int gemmBlocks = (M / 64) * (E_ / 64);   // 64x64 tile per block
  k_gemm<false><<<gemmBlocks, 128, 0, stream>>>(xq,  wq, nullptr, qb, M, E_, E_);
  k_gemm<false><<<gemmBlocks, 128, 0, stream>>>(xkv, wk, nullptr, kb, M, E_, E_);
  k_gemm<false><<<gemmBlocks, 128, 0, stream>>>(xkv, wv, nullptr, vb, M, E_, E_);

  // 3) qk-norm + rope (+ fold sqrt(D) into q)
  k_rms_rope<<<B_ * L_ * H_ / 4, 128, 0, stream>>>(qb, rope_q, q_scale, 11.313708498984761f);
  k_rms_rope<<<B_ * L_ * H_ / 4, 128, 0, stream>>>(kb, rope_k, k_scale, 1.0f);

  // 4) flash attention (bias-added, online softmax)
  k_flash<<<B_ * H_ * (L_ / 64), 128, 0, stream>>>(qb, kb, vb, bias, ab);

  // 5) output projection -> fp32
  k_gemm<true><<<gemmBlocks, 128, 0, stream>>>(ab, wo, out, nullptr, M, E_, E_);
}
